// KBGAT_39805756900141
// MI455X (gfx1250) — compile-verified, matrix operand order checked by reading, not run on hardware
//
#include <hip/hip_runtime.h>

typedef __bf16 bf16;
typedef __attribute__((ext_vector_type(16))) __bf16 v16bf;
typedef __attribute__((ext_vector_type(8)))  __bf16 v8bf;
typedef __attribute__((ext_vector_type(8)))  float  v8f;

// ---------------- problem sizes (compile-time) ----------------
#define N_NODES   16384
#define N_EDGES   131072
#define DIM       512
#define HEADS     8
#define DVAL      64
#define RELD      96            // 64 + 32
#define INDIM     1120          // 2*512 + 96
#define NEG_SLOPE 0.2f
#define EPSV      1e-12f

// ---------------- helpers ----------------
__device__ __forceinline__ float bf2f(bf16 x) { return (float)x; }

// Build a 16-half A/B fragment for v_wmma_f32_16x16x32_bf16 from a row-major
// bf16 row.  Per ISA 7.12.2 (16-bit 16x32 layout): lanes 0-15 hold K=kb..kb+7
// and K=kb+16..kb+23 with kb=0; lanes 16-31 the same with kb=8.
__device__ __forceinline__ v16bf load_frag(const bf16* __restrict__ row, int k0, int kb) {
    v8bf lo = *(const v8bf*)(row + k0 + kb);        // 16B load
    v8bf hi = *(const v8bf*)(row + k0 + kb + 16);   // 16B load
    v16bf r;
#pragma unroll
    for (int i = 0; i < 8; ++i) { r[i] = lo[i]; r[i + 8] = hi[i]; }
    return r;
}

#define WMMA_BF16(A, B, C) \
    __builtin_amdgcn_wmma_f32_16x16x32_bf16(false, (A), false, (B), (short)0, (C), false, false)

// ---------------- 0: out = bias, denom = 0 ----------------
__global__ void init_kernel(float* __restrict__ out, const float* __restrict__ bias,
                            float* __restrict__ denom) {
    long i = (long)blockIdx.x * blockDim.x + threadIdx.x;
    if (i < (long)N_NODES * DIM) out[i] = bias[i & (DIM - 1)];
    if (i < (long)N_NODES * HEADS) denom[i] = 0.0f;
}

// ---------------- 1: fp32 -> bf16 convert ----------------
__global__ void cvt_kernel(const float* __restrict__ in, bf16* __restrict__ o, long n) {
    long i = (long)blockIdx.x * blockDim.x + threadIdx.x;
    if (i < n) o[i] = (bf16)in[i];
}

// ---------------- 2: per-edge relation features (bf16) ----------------
__global__ void rel_kernel(const int* __restrict__ deprel_edge, const int* __restrict__ deparc_edge,
                           const float* __restrict__ deprel_emb, const float* __restrict__ deparc_emb,
                           bf16* __restrict__ rel_bf) {
    long i = (long)blockIdx.x * blockDim.x + threadIdx.x;
    if (i >= (long)N_EDGES * RELD) return;
    int e = (int)(i / RELD), j = (int)(i % RELD);
    float v = (j < 64) ? deprel_emb[(long)deprel_edge[e] * 64 + j]
                       : deparc_emb[(long)deparc_edge[e] * 32 + (j - 64)];
    rel_bf[i] = (bf16)v;
}

// ---------------- 3: WMMA GEMM  h[E,512] = x @ W1^T ----------------
// Register-tiled: each wave computes a 64x64 strip (4 M-tiles x 4 N-tiles,
// 16 accumulators). Per k-step: 4 A-frags + 4 B-frags feed 16 WMMAs
// (16 B/lane/WMMA of fragment traffic). Block = 8 waves = 8 feature groups;
// blockIdx.x = group of 64 edges.
__global__ __launch_bounds__(256)
void gemm_h_kernel(const bf16* __restrict__ inp_bf, const bf16* __restrict__ W1_bf,
                   const bf16* __restrict__ rel_bf, const int* __restrict__ src,
                   const int* __restrict__ tgt, bf16* __restrict__ h_bf) {
    const int lane = threadIdx.x & 31;
    const int ngrp = threadIdx.x >> 5;      // 0..7 : which 64-feature group
    const int mgrp = blockIdx.x;            // 0..E/64-1 : which 64-edge group
    const int m    = lane & 15;
    const int kb   = (lane >> 4) ? 8 : 0;

    // per-M-tile A row base pointers for the three K-segments
    const bf16* a1[4];
    const bf16* a2[4];
    const bf16* a3[4];
#pragma unroll
    for (int mt = 0; mt < 4; ++mt) {
        int e = mgrp * 64 + mt * 16 + m;
        a1[mt] = inp_bf + (long)src[e] * DIM;   // x[:, 0:512]
        a2[mt] = inp_bf + (long)tgt[e] * DIM;   // x[:, 512:1024]
        a3[mt] = rel_bf + (long)e * RELD;       // x[:, 1024:1120]
    }
    const bf16* brow[4];
#pragma unroll
    for (int nt = 0; nt < 4; ++nt)
        brow[nt] = W1_bf + (long)(ngrp * 64 + nt * 16 + m) * INDIM;

    v8f acc[4][4];
#pragma unroll
    for (int mt = 0; mt < 4; ++mt)
#pragma unroll
        for (int nt = 0; nt < 4; ++nt)
#pragma unroll
            for (int i = 0; i < 8; ++i) acc[mt][nt][i] = 0.0f;

    // segment 1: K = 0..511 (src features)
    for (int ks = 0; ks < 16; ++ks) {
        v16bf b[4];
#pragma unroll
        for (int nt = 0; nt < 4; ++nt) b[nt] = load_frag(brow[nt], ks * 32, kb);
#pragma unroll
        for (int mt = 0; mt < 4; ++mt) {
            v16bf a = load_frag(a1[mt], ks * 32, kb);
#pragma unroll
            for (int nt = 0; nt < 4; ++nt) acc[mt][nt] = WMMA_BF16(a, b[nt], acc[mt][nt]);
        }
    }
    // segment 2: K = 512..1023 (tgt features)
    for (int ks = 0; ks < 16; ++ks) {
        v16bf b[4];
#pragma unroll
        for (int nt = 0; nt < 4; ++nt) b[nt] = load_frag(brow[nt], 512 + ks * 32, kb);
#pragma unroll
        for (int mt = 0; mt < 4; ++mt) {
            v16bf a = load_frag(a2[mt], ks * 32, kb);
#pragma unroll
            for (int nt = 0; nt < 4; ++nt) acc[mt][nt] = WMMA_BF16(a, b[nt], acc[mt][nt]);
        }
    }
    // segment 3: K = 1024..1119 (relation features)
    for (int ks = 0; ks < 3; ++ks) {
        v16bf b[4];
#pragma unroll
        for (int nt = 0; nt < 4; ++nt) b[nt] = load_frag(brow[nt], 1024 + ks * 32, kb);
#pragma unroll
        for (int mt = 0; mt < 4; ++mt) {
            v16bf a = load_frag(a3[mt], ks * 32, kb);
#pragma unroll
            for (int nt = 0; nt < 4; ++nt) acc[mt][nt] = WMMA_BF16(a, b[nt], acc[mt][nt]);
        }
    }

    // D layout: lanes 0-15 hold rows M=0..7 (VGPR r -> M=r), lanes 16-31 M=8..15
    const int mb = (lane >> 4) ? 8 : 0;
#pragma unroll
    for (int mt = 0; mt < 4; ++mt)
#pragma unroll
        for (int nt = 0; nt < 4; ++nt) {
            int n = ngrp * 64 + nt * 16 + m;
#pragma unroll
            for (int r = 0; r < 8; ++r)
                h_bf[(long)(mgrp * 64 + mt * 16 + mb + r) * DIM + n] = (bf16)acc[mt][nt][r];
        }
}

// ---------------- 4: scores + per-block max ----------------
__global__ __launch_bounds__(256)
void score_kernel(const bf16* __restrict__ h_bf, const float* __restrict__ W2,
                  float* __restrict__ scores, float* __restrict__ blockmax) {
    __shared__ float sm[256];
    long t = (long)blockIdx.x * 256 + threadIdx.x;     // t < E*H exactly
    int e = (int)(t >> 3), head = (int)(t & 7);
    const bf16* hp = h_bf + (long)e * DIM + head * DVAL;
    const float* w = W2 + head * DVAL;
    float s = 0.0f;
#pragma unroll
    for (int c = 0; c < 8; ++c) {
        v8bf hv = *(const v8bf*)(hp + c * 8);
#pragma unroll
        for (int i = 0; i < 8; ++i) s += bf2f(hv[i]) * w[c * 8 + i];
    }
    s = (s > 0.0f) ? s : NEG_SLOPE * s;   // leaky relu
    scores[t] = s;
    sm[threadIdx.x] = s;
    __syncthreads();
    for (int off = 128; off > 0; off >>= 1) {
        if (threadIdx.x < off) sm[threadIdx.x] = fmaxf(sm[threadIdx.x], sm[threadIdx.x + off]);
        __syncthreads();
    }
    if (threadIdx.x == 0) blockmax[blockIdx.x] = sm[0];
}

// ---------------- 5: reduce block maxes -> global max ----------------
__global__ __launch_bounds__(256)
void reduce_max_kernel(const float* __restrict__ blockmax, int n, float* __restrict__ gmax) {
    __shared__ float sm[256];
    float m = -3.4e38f;
    for (int i = threadIdx.x; i < n; i += 256) m = fmaxf(m, blockmax[i]);
    sm[threadIdx.x] = m;
    __syncthreads();
    for (int off = 128; off > 0; off >>= 1) {
        if (threadIdx.x < off) sm[threadIdx.x] = fmaxf(sm[threadIdx.x], sm[threadIdx.x + off]);
        __syncthreads();
    }
    if (threadIdx.x == 0) *gmax = sm[0];
}

// ---------------- 6: exp + segment-sum denominator ----------------
__global__ __launch_bounds__(256)
void expdenom_kernel(float* __restrict__ scores, const int* __restrict__ tgt,
                     const float* __restrict__ gmax, float* __restrict__ denom) {
    long t = (long)blockIdx.x * 256 + threadIdx.x;     // t < E*H exactly
    int e = (int)(t >> 3), head = (int)(t & 7);
    float es = __expf(scores[t] - *gmax);
    scores[t] = es;                                    // in-place: now exp_scores
    atomicAdd(&denom[(long)tgt[e] * HEADS + head], es);
}

// ---------------- 7: scatter-aggregate ----------------
__global__ __launch_bounds__(256)
void agg_kernel(const bf16* __restrict__ h_bf, const float* __restrict__ exp_scores,
                const float* __restrict__ denom, const int* __restrict__ tgt,
                float* __restrict__ out) {
    long t = (long)blockIdx.x * 256 + threadIdx.x;     // t < E*DIM exactly
    int e = (int)(t >> 9), f = (int)(t & (DIM - 1));
    int head = f >> 6;
    int tg = tgt[e];
    float attn = exp_scores[(long)e * HEADS + head] /
                 (denom[(long)tg * HEADS + head] + EPSV);
    atomicAdd(&out[(long)tg * DIM + f], attn * bf2f(h_bf[t]));
}

// ---------------- launcher ----------------
extern "C" void kernel_launch(void* const* d_in, const int* in_sizes, int n_in,
                              void* d_out, int out_size, void* d_ws, size_t ws_size,
                              hipStream_t stream) {
    const float* inp         = (const float*)d_in[0];
    const int*   deprel_edge = (const int*)  d_in[1];
    const int*   deparc_edge = (const int*)  d_in[2];
    const int*   edge_index  = (const int*)  d_in[3];
    const float* deprel_emb  = (const float*)d_in[4];
    const float* deparc_emb  = (const float*)d_in[5];
    const float* W1          = (const float*)d_in[6];
    const float* W2          = (const float*)d_in[7];
    const float* final_bias  = (const float*)d_in[8];
    float*       out         = (float*)d_out;

    const int* src = edge_index;
    const int* tgt = edge_index + N_EDGES;

    // workspace layout (all offsets 256B-aligned)
    char* ws = (char*)d_ws;
    bf16*  inp_bf = (bf16*)(ws + 0);                        // 16 MB
    bf16*  W1_bf  = (bf16*)(ws + 16777216);                 // 1.09 MB
    bf16*  rel_bf = (bf16*)(ws + 17924096);                 // 24 MB
    bf16*  h_bf   = (bf16*)(ws + 43089920);                 // 128 MB
    float* scores = (float*)(ws + 177307648);               // 4 MB (scores -> exp)
    float* denom  = (float*)(ws + 181501952);               // 0.5 MB
    float* bmax   = (float*)(ws + 182026240);               // 16 KB
    float* gmax   = (float*)(ws + 182042624);               // 4 B

    const long nOut = (long)N_NODES * DIM;                  // 8,388,608
    init_kernel<<<(unsigned)((nOut + 255) / 256), 256, 0, stream>>>(out, final_bias, denom);

    cvt_kernel<<<(unsigned)((nOut + 255) / 256), 256, 0, stream>>>(inp, inp_bf, nOut);
    const long nW1 = (long)(HEADS * DVAL) * INDIM;          // 573,440
    cvt_kernel<<<(unsigned)((nW1 + 255) / 256), 256, 0, stream>>>(W1, W1_bf, nW1);
    const long nRel = (long)N_EDGES * RELD;                 // 12,582,912
    rel_kernel<<<(unsigned)((nRel + 255) / 256), 256, 0, stream>>>(
        deprel_edge, deparc_edge, deprel_emb, deparc_emb, rel_bf);

    gemm_h_kernel<<<N_EDGES / 64, 256, 0, stream>>>(inp_bf, W1_bf, rel_bf, src, tgt, h_bf);

    const int scoreBlocks = (N_EDGES * HEADS) / 256;        // 4096
    score_kernel<<<scoreBlocks, 256, 0, stream>>>(h_bf, W2, scores, bmax);
    reduce_max_kernel<<<1, 256, 0, stream>>>(bmax, scoreBlocks, gmax);
    expdenom_kernel<<<scoreBlocks, 256, 0, stream>>>(scores, tgt, gmax, denom);

    const long nAgg = (long)N_EDGES * DIM;                  // 67,108,864
    agg_kernel<<<(unsigned)(nAgg / 256), 256, 0, stream>>>(h_bf, scores, denom, tgt, out);
}
